// GNN_Het_39702677684849
// MI455X (gfx1250) — matrix operand order repaired and checked by reading, not compiled, not run on hardware
//
#include <hip/hip_runtime.h>

typedef __attribute__((ext_vector_type(16))) _Float16 v16h;
typedef __attribute__((ext_vector_type(8)))  float    v8f;

#define D_FEAT 128   // input / hidden feature width (K per half = 128, concat K = 256)

// ---------------------------------------------------------------------------
// zero a float region (grid-stride)
// ---------------------------------------------------------------------------
__global__ void zero_f32(float* __restrict__ p, long long n) {
  long long i = (long long)blockIdx.x * blockDim.x + threadIdx.x;
  long long s = (long long)gridDim.x * blockDim.x;
  for (; i < n; i += s) p[i] = 0.0f;
}

// ---------------------------------------------------------------------------
// edge scatter: one wave (32 lanes) per edge, each lane moves 4 floats.
// agg[dst] += feat[src] (row-contiguous float4 gather, native f32 atomics).
// Lane 0 optionally bumps the degree counter.
// ---------------------------------------------------------------------------
__global__ void sage_scatter(const long long* __restrict__ ei,   // [2,E] int64
                             const float* __restrict__ feat,     // [N,128]
                             float* __restrict__ agg,             // [N,128]
                             float* __restrict__ cnt,             // [N]
                             int nE, int addCnt) {
  long long t = (long long)blockIdx.x * blockDim.x + threadIdx.x;
  int e    = (int)(t >> 5);
  int lane = (int)(t & 31);
  if (e >= nE) return;
  long long src = ei[e];
  long long dst = ei[(long long)nE + e];
  const float4 v = *(const float4*)(feat + src * D_FEAT + lane * 4);
  float* a = agg + dst * D_FEAT + lane * 4;
  unsafeAtomicAdd(a + 0, v.x);
  unsafeAtomicAdd(a + 1, v.y);
  unsafeAtomicAdd(a + 2, v.z);
  unsafeAtomicAdd(a + 3, v.w);
  if (addCnt && lane == 0) unsafeAtomicAdd(cnt + dst, 1.0f);
}

// ---------------------------------------------------------------------------
// inv[i] = 1 / max(cnt[i], 1)
// ---------------------------------------------------------------------------
__global__ void inv_cnt(const float* __restrict__ cnt, float* __restrict__ inv, int n) {
  int i = blockIdx.x * blockDim.x + threadIdx.x;
  if (i < n) inv[i] = 1.0f / fmaxf(cnt[i], 1.0f);
}

// ---------------------------------------------------------------------------
// pack concatenated weights [W_l ; W_r] (256 x ncols, fp32) into per-lane
// contiguous f16 B-fragments for v_wmma_f32_16x16x32_f16.
// B striping (ISA dense 16-bit order): lanes 0-15 -> K = kb*32 + {0..7,16..23},
// lanes 16-31 -> K = kb*32 + {8..15,24..31}; N = nb*16 + (lane&15);
// element e (0..15) walks that K set in ascending order.
// dst index = (((kb*NB + nb)*32) + lane)*16 + e
// ---------------------------------------------------------------------------
__global__ void pack_w(const float* __restrict__ Wl, const float* __restrict__ Wr,
                       _Float16* __restrict__ dst, int NB) {
  int idx   = blockIdx.x * blockDim.x + threadIdx.x;
  int total = 8 * NB * 32 * 16;
  if (idx >= total) return;
  int e    = idx & 15;
  int lane = (idx >> 4) & 31;
  int rest = idx >> 9;           // kb*NB + nb
  int nb   = rest % NB;
  int kb   = rest / NB;
  int k    = kb * 32 + ((lane >= 16) ? 8 : 0) + (e & 7) + ((e >= 8) ? 16 : 0);
  int n    = nb * 16 + (lane & 15);
  int nc   = NB * 16;
  float v  = (k < D_FEAT) ? Wl[k * nc + n] : Wr[(k - D_FEAT) * nc + n];
  dst[idx] = (_Float16)v;
}

// ---------------------------------------------------------------------------
// Fused SAGE layer GEMM:
//   out = act( [agg*inv | self] (N x 256) @ Wpacked (256 x NB*16) + bias )
// One wave per 16-row M tile; NB accumulators (NB*8 VGPRs) in registers.
// A fragment per ISA 16-bit A layout: lane<16 row=lane, K chunks {0..7,16..23};
// lane>=16 same row set, K chunks {8..15,24..31}.  Mean division fused as a
// scale on the agg half of K.  kb 0..3 -> agg, kb 4..7 -> self features.
// ---------------------------------------------------------------------------
template <int NB, bool RELU>
__global__ void __launch_bounds__(256) sage_gemm(
    const float* __restrict__ agg, const float* __restrict__ self,
    const float* __restrict__ inv, const _Float16* __restrict__ wp,
    const float* __restrict__ bias, float* __restrict__ out,
    int nNodes, int mtiles) {
  int wave = (int)(((long long)blockIdx.x * blockDim.x + threadIdx.x) >> 5);
  int lane = threadIdx.x & 31;
  if (wave >= mtiles) return;           // wave-uniform exit: EXEC stays all-1s

  const int half  = lane >> 4;          // 0 or 1
  const int mrow  = lane & 15;
  const int kbase = half * 8;
  const int ncols = NB * 16;

  long long row  = (long long)wave * 16 + mrow;
  long long rowc = row < nNodes ? row : (long long)nNodes - 1;   // clamp loads
  const float scaleAgg = inv[rowc];
  const float* aggRow  = agg  + rowc * D_FEAT;
  const float* selfRow = self + rowc * D_FEAT;

  v8f acc[NB];
#pragma unroll
  for (int nb = 0; nb < NB; ++nb) {
    float b = bias[nb * 16 + mrow];
#pragma unroll
    for (int r = 0; r < 8; ++r) acc[nb][r] = b;
  }

#pragma unroll
  for (int kb = 0; kb < 8; ++kb) {
    const float* srcRow = (kb < 4) ? aggRow : selfRow;
    const float  scale  = (kb < 4) ? scaleAgg : 1.0f;
    const int    kcol   = (kb & 3) * 32 + kbase;
    float4 a0 = *(const float4*)(srcRow + kcol);
    float4 a1 = *(const float4*)(srcRow + kcol + 4);
    float4 a2 = *(const float4*)(srcRow + kcol + 16);
    float4 a3 = *(const float4*)(srcRow + kcol + 20);
    v16h A;
    A[0]  = (_Float16)(a0.x * scale); A[1]  = (_Float16)(a0.y * scale);
    A[2]  = (_Float16)(a0.z * scale); A[3]  = (_Float16)(a0.w * scale);
    A[4]  = (_Float16)(a1.x * scale); A[5]  = (_Float16)(a1.y * scale);
    A[6]  = (_Float16)(a1.z * scale); A[7]  = (_Float16)(a1.w * scale);
    A[8]  = (_Float16)(a2.x * scale); A[9]  = (_Float16)(a2.y * scale);
    A[10] = (_Float16)(a2.z * scale); A[11] = (_Float16)(a2.w * scale);
    A[12] = (_Float16)(a3.x * scale); A[13] = (_Float16)(a3.y * scale);
    A[14] = (_Float16)(a3.z * scale); A[15] = (_Float16)(a3.w * scale);
#pragma unroll
    for (int nb = 0; nb < NB; ++nb) {
      v16h B = *(const v16h*)(wp + (((kb * NB + nb) * 32) + lane) * 16);
      acc[nb] = __builtin_amdgcn_wmma_f32_16x16x32_f16(
          false, A, false, B, (short)0, acc[nb], false, false);
    }
  }

  // C/D layout: col = nb*16 + (lane&15); VGPR r -> row m0 + half*8 + r
#pragma unroll
  for (int nb = 0; nb < NB; ++nb) {
    int col = nb * 16 + mrow;
#pragma unroll
    for (int r = 0; r < 8; ++r) {
      long long orow = (long long)wave * 16 + half * 8 + r;
      if (orow < nNodes) {
        float v = acc[nb][r];
        if (RELU) v = fmaxf(v, 0.0f);
        out[orow * ncols + col] = v;
      }
    }
  }
}

// ---------------------------------------------------------------------------
// launch
// ---------------------------------------------------------------------------
extern "C" void kernel_launch(void* const* d_in, const int* in_sizes, int n_in,
                              void* d_out, int out_size, void* d_ws, size_t ws_size,
                              hipStream_t stream) {
  const float*     x   = (const float*)d_in[0];
  const long long* ei  = (const long long*)d_in[1];   // int64 edge_index [2,E]
  const float*     Wl1 = (const float*)d_in[2];
  const float*     Wr1 = (const float*)d_in[3];
  const float*     b1  = (const float*)d_in[4];
  const float*     Wl2 = (const float*)d_in[5];
  const float*     Wr2 = (const float*)d_in[6];
  const float*     b2  = (const float*)d_in[7];

  const int nNodes = in_sizes[0] / D_FEAT;      // 100000
  const int nE     = in_sizes[1] / 2;           // 1600000
  const int mtiles = (nNodes + 15) / 16;        // 6250

  // workspace layout: agg | cnt | h | inv | w1p | w2p
  float* agg = (float*)d_ws;
  float* cnt = agg + (size_t)nNodes * D_FEAT;
  float* h   = cnt + nNodes;
  float* inv = h + (size_t)nNodes * D_FEAT;
  _Float16* w1p = (_Float16*)(inv + nNodes);
  _Float16* w2p = w1p + 8 * 8 * 32 * 16;

  const int T = 256;
  const long long scatterThreads = (long long)nE * 32;
  const int scatterBlocks = (int)((scatterThreads + T - 1) / T);
  const int gemmBlocks    = (mtiles * 32 + T - 1) / T;

  // layer 1
  zero_f32<<<2048, T, 0, stream>>>(agg, (long long)nNodes * D_FEAT + nNodes);
  pack_w<<<(8 * 8 * 32 * 16 + T - 1) / T, T, 0, stream>>>(Wl1, Wr1, w1p, 8);
  pack_w<<<(8 * 4 * 32 * 16 + T - 1) / T, T, 0, stream>>>(Wl2, Wr2, w2p, 4);
  sage_scatter<<<scatterBlocks, T, 0, stream>>>(ei, x, agg, cnt, nE, 1);
  inv_cnt<<<(nNodes + T - 1) / T, T, 0, stream>>>(cnt, inv, nNodes);
  sage_gemm<8, true><<<gemmBlocks, T, 0, stream>>>(agg, x, inv, w1p, b1, h,
                                                   nNodes, mtiles);
  // layer 2 (reuse agg buffer)
  zero_f32<<<2048, T, 0, stream>>>(agg, (long long)nNodes * D_FEAT);
  sage_scatter<<<scatterBlocks, T, 0, stream>>>(ei, h, agg, cnt, nE, 0);
  sage_gemm<4, false><<<gemmBlocks, T, 0, stream>>>(agg, h, inv, w2p, b2,
                                                    (float*)d_out, nNodes, mtiles);
}
